// DualWeightAttention_68436008894973
// MI455X (gfx1250) — compile-verified, matrix operand order checked by reading, not run on hardware
//
#include <hip/hip_runtime.h>

// ---------------------------------------------------------------------------
// DualWeightAttention for MI455X (gfx1250, wave32, WMMA bf16 16x16x32,
// async global->LDS staging, double-buffered).
// B=2, S=2048, H=2048, nh=16, hd=128
// ---------------------------------------------------------------------------

#define SEQ   2048
#define HDIM  2048
#define NH    16
#define HD    128
#define BATCH 2
#define NTOK  (BATCH * SEQ)      // 4096
#define CK    32                 // attention key-chunk

typedef __attribute__((ext_vector_type(16))) __bf16 v16bf;
typedef __attribute__((ext_vector_type(8)))  float  v8f;
typedef __attribute__((ext_vector_type(4)))  float  f32x4;
typedef __attribute__((ext_vector_type(2)))  unsigned int u32x2;
typedef __attribute__((ext_vector_type(4)))  int    v4i_t;

// ---- WMMA wrapper: D = A(16x32 bf16) * B(32x16 bf16) + C(16x16 f32) -------
__device__ __forceinline__ v8f wmma_bf16(v16bf a, v16bf b, v8f c) {
  return __builtin_amdgcn_wmma_f32_16x16x32_bf16(
      false, a, false, b, (short)0, c, false, false);
}

// ---- async global -> LDS 16B copy (CDNA5 GLOBAL_LOAD_ASYNC_TO_LDS_B128) ---
// Builtin signature (from hipcc diagnostic): (v4i AS1*, v4i AS3*, imm, imm).
typedef __attribute__((address_space(1))) v4i_t* gas1_v4i;  // ptr to AS1 v4i
typedef __attribute__((address_space(3))) v4i_t* las3_v4i;  // ptr to AS3 v4i

__device__ __forceinline__ void async_copy_b128(void* lds_dst, const void* gsrc) {
#if __has_builtin(__builtin_amdgcn_global_load_async_to_lds_b128)
  __builtin_amdgcn_global_load_async_to_lds_b128(
      (gas1_v4i)gsrc, (las3_v4i)lds_dst, 0, 0);
#else
  *(f32x4*)lds_dst = *(const f32x4*)gsrc;   // sync fallback
#endif
}

template <int N>
__device__ __forceinline__ void wait_async() {
#if __has_builtin(__builtin_amdgcn_s_wait_asynccnt)
  __builtin_amdgcn_s_wait_asynccnt(N);
#endif
}

// ---- Fragment load for a matrix row-major along the K (contraction) dim ---
// CDNA5 16-bit layout: lane l<16  -> row=row0+l,    K = {k0..k0+7, k0+16..k0+23}
//                      lane l>=16 -> row=row0+l-16, K = {k0+8..k0+15, k0+24..k0+31}
template <typename PT>
__device__ __forceinline__ v16bf load_frag(const PT* base, int ld, int row0,
                                           int k0, int lane) {
  int r  = row0 + (lane & 15);
  int kk = k0 + ((lane >> 4) << 3);
  const PT* p = base + (size_t)r * ld + kk;
  union { v16bf v; f32x4 q[2]; } u;
  u.q[0] = *(const f32x4*)(p);
  u.q[1] = *(const f32x4*)(p + 16);
  return u.v;
}

// ---------------------------------------------------------------------------
// f32 -> bf16 bulk convert (4 elems / thread)
// ---------------------------------------------------------------------------
__global__ __launch_bounds__(256) void k_cvt4(const f32x4* __restrict__ in,
                                              u32x2* __restrict__ out, int n4) {
  int i = blockIdx.x * blockDim.x + threadIdx.x;
  if (i < n4) {
    f32x4 f = in[i];
    union { __bf16 h[4]; u32x2 u; } r;
    r.h[0] = (__bf16)f.x; r.h[1] = (__bf16)f.y;
    r.h[2] = (__bf16)f.z; r.h[3] = (__bf16)f.w;
    out[i] = r.u;
  }
}

// ---------------------------------------------------------------------------
// Fused QKV projection. Wave computes a 32x32 tile of Q, K and V
// (12 WMMAs per k-step, A fragments shared 6x). A-tile (32 rows, shared by
// all 8 waves) is staged in LDS via async copies, double buffered.
// Q,K stored [B,nh,S,hd] (Q pre-scaled by 1/sqrt(hd)); V stored transposed.
// Grid: (NTOK/32, HDIM/(32*8)), block 256.
// ---------------------------------------------------------------------------
__global__ __launch_bounds__(256) void k_proj_qkv(
    const __bf16* __restrict__ Xb,  const __bf16* __restrict__ Wqb,
    const __bf16* __restrict__ Wkb, const __bf16* __restrict__ Wvb,
    __bf16* __restrict__ Qb, __bf16* __restrict__ Kb, __bf16* __restrict__ Vtb) {
  __shared__ __align__(16) __bf16 At[2][32 * 32];   // 2 KB per buffer

  int tid  = threadIdx.x;
  int lane = tid & 31;
  int w    = tid >> 5;
  int m0   = blockIdx.x * 32;
  int n0   = (blockIdx.y * 8 + w) * 32;

  v8f acc[3][4];
#pragma unroll
  for (int i = 0; i < 3; ++i)
#pragma unroll
    for (int j = 0; j < 4; ++j) acc[i][j] = (v8f){0,0,0,0,0,0,0,0};

  auto stageA = [&](int k0, int buf) {
    if (tid < 128) {                       // 32 rows x 4 x 16B segments
      int row = tid >> 2, seg = tid & 3;
      async_copy_b128(&At[buf][row * 32 + seg * 8],
                      Xb + (size_t)(m0 + row) * HDIM + k0 + seg * 8);
    }
  };

  const __bf16* Wp[3] = {Wqb, Wkb, Wvb};
  stageA(0, 0);
  for (int k0 = 0, c = 0; k0 < HDIM; k0 += 32, ++c) {
    int buf = c & 1;
    if (k0 + 32 < HDIM) { stageA(k0 + 32, buf ^ 1); wait_async<1>(); }
    else                { wait_async<0>(); }
    __syncthreads();

    v16bf a0 = load_frag((const __bf16*)At[buf], 32, 0,  0, lane);
    v16bf a1 = load_frag((const __bf16*)At[buf], 32, 16, 0, lane);
#pragma unroll
    for (int ww = 0; ww < 3; ++ww) {
      v16bf b0 = load_frag(Wp[ww], HDIM, n0,      k0, lane);
      v16bf b1 = load_frag(Wp[ww], HDIM, n0 + 16, k0, lane);
      acc[ww][0] = wmma_bf16(a0, b0, acc[ww][0]);
      acc[ww][1] = wmma_bf16(a0, b1, acc[ww][1]);
      acc[ww][2] = wmma_bf16(a1, b0, acc[ww][2]);
      acc[ww][3] = wmma_bf16(a1, b1, acc[ww][3]);
    }
    __syncthreads();
  }

  const float qscale = 0.08838834764831845f;  // 1/sqrt(128)
#pragma unroll
  for (int mi = 0; mi < 2; ++mi)
#pragma unroll
    for (int ni = 0; ni < 2; ++ni) {
      int mbase = m0 + mi * 16 + ((lane >> 4) << 3);
      int n     = n0 + ni * 16 + (lane & 15);
      int hN = n >> 7, d = n & 127;
      v8f aq = acc[0][mi * 2 + ni];
      v8f ak = acc[1][mi * 2 + ni];
      v8f av = acc[2][mi * 2 + ni];
#pragma unroll
      for (int i = 0; i < 8; ++i) {
        int t = mbase + i;
        int b = t >> 11, s = t & (SEQ - 1);
        size_t qk = (((size_t)(b * NH + hN)) * SEQ + s) * HD + d;
        Qb[qk] = (__bf16)(aq[i] * qscale);
        Kb[qk] = (__bf16)ak[i];
        Vtb[(((size_t)(b * NH + hN)) * HD + d) * SEQ + s] = (__bf16)av[i];
      }
    }
}

// ---------------------------------------------------------------------------
// Flash attention. All 8 waves share (b,h); K/V chunks (32 keys) are staged
// into LDS with async copies (double buffered), consumed by every wave.
// Each wave owns 16 query rows x full hd with online softmax.
// Grid: (SEQ/16/8, B*NH), block 256.
// ---------------------------------------------------------------------------
__global__ __launch_bounds__(256) void k_attn(
    const __bf16* __restrict__ Qb, const __bf16* __restrict__ Kb,
    const __bf16* __restrict__ Vtb, const float* __restrict__ mask,
    __bf16* __restrict__ AOb) {
  __shared__ __align__(16) __bf16 Kt[2][CK * HD];   // 8 KB x 2
  __shared__ __align__(16) __bf16 Vt[2][HD * CK];   // 8 KB x 2
  __shared__ __align__(16) __bf16 Pt[8][16 * CK];   // 1 KB per wave

  int tid  = threadIdx.x;
  int lane = tid & 31;
  int w    = tid >> 5;
  int bh   = blockIdx.y;                 // b*NH + h
  int b    = bh >> 4, h = bh & 15;
  int q0   = (blockIdx.x * 8 + w) * 16;

  const __bf16* Qp = Qb  + (size_t)bh * SEQ * HD;
  const __bf16* Kp = Kb  + (size_t)bh * SEQ * HD;
  const __bf16* Vp = Vtb + (size_t)bh * HD * SEQ;   // [hd, S]
  const float*  mp = mask + (size_t)b * SEQ * SEQ;  // [S, S]

  // Q fragments: 16 rows x 128 = 4 x (16x32)
  v16bf qf[4];
#pragma unroll
  for (int t = 0; t < 4; ++t) qf[t] = load_frag(Qp, HD, q0, t * 32, lane);

  v8f oacc[8];
#pragma unroll
  for (int j = 0; j < 8; ++j) oacc[j] = (v8f){0,0,0,0,0,0,0,0};
  float rowm[8], rowl[8];
#pragma unroll
  for (int i = 0; i < 8; ++i) { rowm[i] = -3.0e38f; rowl[i] = 0.0f; }

  int mrow = q0 + ((lane >> 4) << 3);
  int ncol = lane & 15;

  // async stage of K (contiguous 8KB) and V (128 rows x 64B) for chunk c
  auto stage = [&](int c, int buf) {
    const __bf16* kg = Kp + (size_t)c * CK * HD;
#pragma unroll
    for (int i = 0; i < 2; ++i) {
      int idx = tid + i * 256;                    // 0..511 x 16B
      async_copy_b128(&Kt[buf][idx * 8], kg + idx * 8);
    }
    const __bf16* vg = Vp + c * CK;
#pragma unroll
    for (int i = 0; i < 2; ++i) {
      int idx = tid + i * 256;
      int row = idx >> 2, seg = idx & 3;          // 128 rows x 4 x 16B
      async_copy_b128(&Vt[buf][row * CK + seg * 8],
                      vg + (size_t)row * SEQ + seg * 8);
    }
  };

  const int NCHUNK = SEQ / CK;   // 64
  stage(0, 0);
  for (int c = 0; c < NCHUNK; ++c) {
    int buf = c & 1;
    if (c + 1 < NCHUNK) { stage(c + 1, buf ^ 1); wait_async<4>(); }
    else                { wait_async<0>(); }
    __syncthreads();
    int k0 = c * CK;

    // ---- scores: 16q x 32k, 2 n-tiles x 4 hd-steps = 8 WMMAs (from LDS) --
    v8f sacc[2];
#pragma unroll
    for (int j = 0; j < 2; ++j) sacc[j] = (v8f){0,0,0,0,0,0,0,0};
#pragma unroll
    for (int j = 0; j < 2; ++j)
#pragma unroll
      for (int t = 0; t < 4; ++t) {
        v16bf kf = load_frag((const __bf16*)Kt[buf], HD, j * 16, t * 32, lane);
        sacc[j] = wmma_bf16(qf[t], kf, sacc[j]);
      }

    // ---- additive mask + running row max ---------------------------------
    float newm[8];
#pragma unroll
    for (int i = 0; i < 8; ++i) newm[i] = rowm[i];
#pragma unroll
    for (int j = 0; j < 2; ++j)
#pragma unroll
      for (int i = 0; i < 8; ++i) {
        float sv = sacc[j][i] +
                   mp[(size_t)(mrow + i) * SEQ + (k0 + j * 16 + ncol)];
        sacc[j][i] = sv;
        newm[i] = fmaxf(newm[i], sv);
      }
#pragma unroll
    for (int i = 0; i < 8; ++i)
#pragma unroll
      for (int off = 1; off < 16; off <<= 1)
        newm[i] = fmaxf(newm[i], __shfl_xor(newm[i], off, 32));

    // ---- exponentiate, rescale, stash P (C-layout -> row-major bf16) -----
    float escale[8], psum[8];
#pragma unroll
    for (int i = 0; i < 8; ++i) {
      escale[i] = __expf(rowm[i] - newm[i]);
      rowm[i]   = newm[i];
      psum[i]   = 0.0f;
    }
#pragma unroll
    for (int j = 0; j < 2; ++j)
#pragma unroll
      for (int i = 0; i < 8; ++i) {
        float p = __expf(sacc[j][i] - newm[i]);
        psum[i] += p;
        Pt[w][(((lane >> 4) << 3) + i) * CK + j * 16 + ncol] = (__bf16)p;
      }
#pragma unroll
    for (int i = 0; i < 8; ++i) {
#pragma unroll
      for (int off = 1; off < 16; off <<= 1)
        psum[i] += __shfl_xor(psum[i], off, 32);
      rowl[i] = rowl[i] * escale[i] + psum[i];
    }
#pragma unroll
    for (int j = 0; j < 8; ++j)
#pragma unroll
      for (int i = 0; i < 8; ++i) oacc[j][i] *= escale[i];

    // ---- PV: P (16x32) as A-operand, V tile from LDS; 8 WMMAs ------------
    // DS ops are in-order within a wave: pa read sees Pt stores above.
    v16bf pa = load_frag((const __bf16*)Pt[w], CK, 0, 0, lane);
#pragma unroll
    for (int j = 0; j < 8; ++j) {
      v16bf vf = load_frag((const __bf16*)Vt[buf], CK, j * 16, 0, lane);
      oacc[j] = wmma_bf16(pa, vf, oacc[j]);
    }
    __syncthreads();
  }

  // ---- normalize and store merged-head output [B,S,H] bf16 ---------------
#pragma unroll
  for (int i = 0; i < 8; ++i) rowl[i] = 1.0f / rowl[i];
#pragma unroll
  for (int j = 0; j < 8; ++j)
#pragma unroll
    for (int i = 0; i < 8; ++i) {
      int s = mrow + i;
      AOb[((size_t)(b * SEQ + s)) * HDIM + h * HD + j * 16 + ncol] =
          (__bf16)(oacc[j][i] * rowl[i]);
    }
}

// ---------------------------------------------------------------------------
// Output projection: out = AO @ Wo^T, f32 result. Same LDS-staged A-tile and
// 32x32 register blocking (4 WMMAs / 2 global fragment loads per k-step).
// ---------------------------------------------------------------------------
__global__ __launch_bounds__(256) void k_out_gemm(
    const __bf16* __restrict__ AOb, const __bf16* __restrict__ Wob,
    float* __restrict__ out) {
  __shared__ __align__(16) __bf16 At[2][32 * 32];

  int tid  = threadIdx.x;
  int lane = tid & 31;
  int w    = tid >> 5;
  int m0   = blockIdx.x * 32;
  int n0   = (blockIdx.y * 8 + w) * 32;

  v8f acc[4];
#pragma unroll
  for (int j = 0; j < 4; ++j) acc[j] = (v8f){0,0,0,0,0,0,0,0};

  auto stageA = [&](int k0, int buf) {
    if (tid < 128) {
      int row = tid >> 2, seg = tid & 3;
      async_copy_b128(&At[buf][row * 32 + seg * 8],
                      AOb + (size_t)(m0 + row) * HDIM + k0 + seg * 8);
    }
  };

  stageA(0, 0);
  for (int k0 = 0, c = 0; k0 < HDIM; k0 += 32, ++c) {
    int buf = c & 1;
    if (k0 + 32 < HDIM) { stageA(k0 + 32, buf ^ 1); wait_async<1>(); }
    else                { wait_async<0>(); }
    __syncthreads();

    v16bf a0 = load_frag((const __bf16*)At[buf], 32, 0,  0, lane);
    v16bf a1 = load_frag((const __bf16*)At[buf], 32, 16, 0, lane);
    v16bf b0 = load_frag(Wob, HDIM, n0,      k0, lane);
    v16bf b1 = load_frag(Wob, HDIM, n0 + 16, k0, lane);
    acc[0] = wmma_bf16(a0, b0, acc[0]);
    acc[1] = wmma_bf16(a0, b1, acc[1]);
    acc[2] = wmma_bf16(a1, b0, acc[2]);
    acc[3] = wmma_bf16(a1, b1, acc[3]);
    __syncthreads();
  }

#pragma unroll
  for (int mi = 0; mi < 2; ++mi)
#pragma unroll
    for (int ni = 0; ni < 2; ++ni) {
      int m = m0 + mi * 16 + ((lane >> 4) << 3);
      int n = n0 + ni * 16 + (lane & 15);
      v8f a = acc[mi * 2 + ni];
#pragma unroll
      for (int i = 0; i < 8; ++i)
        out[(size_t)(m + i) * HDIM + n] = a[i];
    }
}

// ---------------------------------------------------------------------------
extern "C" void kernel_launch(void* const* d_in, const int* in_sizes, int n_in,
                              void* d_out, int out_size, void* d_ws, size_t ws_size,
                              hipStream_t stream) {
  const float* hs   = (const float*)d_in[0];
  const float* mask = (const float*)d_in[1];
  const float* Wq   = (const float*)d_in[2];
  const float* Wk   = (const float*)d_in[3];
  const float* Wv   = (const float*)d_in[4];
  const float* Wo   = (const float*)d_in[5];
  float* out = (float*)d_out;

  char* ws = (char*)d_ws;
  const size_t SZ_X = (size_t)NTOK * HDIM * 2;
  const size_t SZ_W = (size_t)HDIM * HDIM * 2;
  const size_t SZ_Q = (size_t)NTOK * HD * NH * 2;
  __bf16* Xb  = (__bf16*)(ws);
  __bf16* Wqb = (__bf16*)(ws + SZ_X);
  __bf16* Wkb = (__bf16*)(ws + SZ_X + SZ_W);
  __bf16* Wvb = (__bf16*)(ws + SZ_X + 2 * SZ_W);
  __bf16* Wob = (__bf16*)(ws + SZ_X + 3 * SZ_W);
  __bf16* Qb  = (__bf16*)(ws + SZ_X + 4 * SZ_W);
  __bf16* Kb  = (__bf16*)(ws + SZ_X + 4 * SZ_W + SZ_Q);
  __bf16* Vtb = (__bf16*)(ws + SZ_X + 4 * SZ_W + 2 * SZ_Q);
  __bf16* AOb = Xb;  // X dead after projections; reuse for attention output

  int n4 = (NTOK * HDIM) / 4;
  int w4 = (HDIM * HDIM) / 4;
  k_cvt4<<<(n4 + 255) / 256, 256, 0, stream>>>((const f32x4*)hs, (u32x2*)Xb, n4);
  k_cvt4<<<(w4 + 255) / 256, 256, 0, stream>>>((const f32x4*)Wq, (u32x2*)Wqb, w4);
  k_cvt4<<<(w4 + 255) / 256, 256, 0, stream>>>((const f32x4*)Wk, (u32x2*)Wkb, w4);
  k_cvt4<<<(w4 + 255) / 256, 256, 0, stream>>>((const f32x4*)Wv, (u32x2*)Wvb, w4);
  k_cvt4<<<(w4 + 255) / 256, 256, 0, stream>>>((const f32x4*)Wo, (u32x2*)Wob, w4);

  k_proj_qkv<<<dim3(NTOK / 32, HDIM / 32 / 8), 256, 0, stream>>>(
      Xb, Wqb, Wkb, Wvb, Qb, Kb, Vtb);

  k_attn<<<dim3(SEQ / 16 / 8, BATCH * NH), 256, 0, stream>>>(
      Qb, Kb, Vtb, mask, AOb);

  k_out_gemm<<<dim3(NTOK / 32, HDIM / 32 / 8), 256, 0, stream>>>(AOb, Wob, out);
}
